// HybridGemmaDiT_592705486940
// MI455X (gfx1250) — compile-verified, matrix operand order checked by reading, not run on hardware
//
#include <hip/hip_runtime.h>
#include <hip/hip_bf16.h>
#include <math.h>

typedef __attribute__((ext_vector_type(16))) __bf16 v16bf;
typedef __attribute__((ext_vector_type(2)))  __bf16 bf16x2;
typedef __attribute__((ext_vector_type(8)))  float  v8f;

#define SEQ 2048
#define DIM 1024
#define NH  16
#define HD  64
#define NE  8
#define HID 4096

// ---------------------------------------------------------------------------
// bf16-WMMA GEMM: C[M,N] = A[M,K] @ B[K,N], fp32 in/out, bf16 compute.
// Block: 256 threads (8 waves), tile 64(m) x 128(n); wave = 16(m) x 64(n)
//   -> 4 v_wmma_f32_16x16x32_bf16 per wave per K-step (8 for MODE 1).
// Double-buffered LDS (ping-pong): global loads of K-tile i+1 are batched into
// registers and overlap the WMMAs consuming K-tile i; fp32->bf16 staged with
// paired v_cvt_pk_bf16_f32 + ds_store_b32.
// MODE 0: C = A*B (+bias if bias!=nullptr)
// MODE 1: C = silu(A*B[:,n]) * (A*B[:,n+dualOff])   (fused SwiGLU)
// MODE 2: C += (A*B) * scale[m*scale_stride]        (expert accumulate)
// ---------------------------------------------------------------------------
template<int MODE>
__global__ __launch_bounds__(256)
void gemm_bf16wmma(const float* __restrict__ A, const float* __restrict__ B,
                   float* __restrict__ C, const float* __restrict__ bias,
                   const float* __restrict__ scale, int scale_stride,
                   int M, int N, int K, int lda, int ldb, int ldc, int dualOff)
{
    __shared__ __bf16 As[2][64][32];     // [buf][m][k]
    __shared__ __bf16 Bs[2][128][32];    // [buf][n][k]  (transposed)
    __shared__ __bf16 Bs2[2][128][32];   // second half for SwiGLU

    const int tid  = threadIdx.x;
    const int lane = tid & 31;
    const int w    = tid >> 5;
    const int wm   = w & 3;           // 0..3  m sub-tile
    const int wn   = w >> 2;          // 0..1  n half (64 wide)
    const int m0   = blockIdx.y * 64;
    const int n0   = blockIdx.x * 128;

    v8f acc[4]  = {{}, {}, {}, {}};
    v8f accB[4] = {{}, {}, {}, {}};

    const int hl   = lane >> 4;
    const int mrow = wm * 16 + (lane & 15);

    // per-thread staging registers for one K-tile
    float2 ra[4];
    float  rb0[8], rb1[8], rc0[8], rc1[8];

    // batched global loads of tile at k0 (single wait, clause-friendly)
    auto gload = [&](int k0) {
#pragma unroll
        for (int p = 0; p < 4; ++p) {
            int idx = p * 256 + tid;
            int r = idx >> 4, cp = idx & 15;
            ra[p] = *(const float2*)(A + (size_t)(m0 + r) * lda + k0 + 2 * cp);
        }
#pragma unroll
        for (int p = 0; p < 8; ++p) {
            int idx = p * 256 + tid;
            int n = idx >> 4, kp = idx & 15;
            const float* bp = B + (size_t)(k0 + 2 * kp) * ldb + n0 + n;
            rb0[p] = bp[0];
            rb1[p] = bp[ldb];
            if constexpr (MODE == 1) {
                rc0[p] = bp[dualOff];
                rc1[p] = bp[dualOff + ldb];
            }
        }
        if (k0 + 64 < K) {   // keep K-tile i+2 moving toward L2
            __builtin_prefetch(A + (size_t)(m0 + (tid & 63)) * lda + k0 + 64, 0, 1);
            __builtin_prefetch(B + (size_t)(k0 + 64 + (tid & 31)) * ldb + n0 + (tid >> 1), 0, 1);
        }
    };
    // convert + store staged tile into LDS buffer `buf`
    auto sstore = [&](int buf) {
#pragma unroll
        for (int p = 0; p < 4; ++p) {
            int idx = p * 256 + tid;
            int r = idx >> 4, cp = idx & 15;
            bf16x2 v; v[0] = (__bf16)ra[p].x; v[1] = (__bf16)ra[p].y;
            *(bf16x2*)&As[buf][r][2 * cp] = v;
        }
#pragma unroll
        for (int p = 0; p < 8; ++p) {
            int idx = p * 256 + tid;
            int n = idx >> 4, kp = idx & 15;
            bf16x2 v; v[0] = (__bf16)rb0[p]; v[1] = (__bf16)rb1[p];
            *(bf16x2*)&Bs[buf][n][2 * kp] = v;
            if constexpr (MODE == 1) {
                bf16x2 v2; v2[0] = (__bf16)rc0[p]; v2[1] = (__bf16)rc1[p];
                *(bf16x2*)&Bs2[buf][n][2 * kp] = v2;
            }
        }
    };
    // WMMA consume of LDS buffer `buf` (ISA 7.12.2 16-bit layouts)
    auto compute = [&](int buf) {
        v16bf a;
#pragma unroll
        for (int e = 0; e < 16; ++e) {
            int kk = (e < 8) ? (e + 8 * hl) : (e + 8 + 8 * hl);
            a[e] = As[buf][mrow][kk];
        }
#pragma unroll
        for (int t = 0; t < 4; ++t) {
            int ncol = wn * 64 + t * 16 + (lane & 15);
            v16bf b;
#pragma unroll
            for (int e = 0; e < 16; ++e) {
                int kk = (e < 8) ? (e + 8 * hl) : (e + 8 + 8 * hl);
                b[e] = Bs[buf][ncol][kk];
            }
            acc[t] = __builtin_amdgcn_wmma_f32_16x16x32_bf16(
                false, a, false, b, (short)0, acc[t], false, false);
            if constexpr (MODE == 1) {
                v16bf b2;
#pragma unroll
                for (int e = 0; e < 16; ++e) {
                    int kk = (e < 8) ? (e + 8 * hl) : (e + 8 + 8 * hl);
                    b2[e] = Bs2[buf][ncol][kk];
                }
                accB[t] = __builtin_amdgcn_wmma_f32_16x16x32_bf16(
                    false, a, false, b2, (short)0, accB[t], false, false);
            }
        }
    };

    // ---- software-pipelined main loop ----
    gload(0);
    sstore(0);
    int buf = 0;
    for (int k0 = 32; k0 < K; k0 += 32) {
        gload(k0);                 // global loads for tile i+1 in flight
        __syncthreads();           // tile i visible in LDS[buf]
        compute(buf);              // WMMAs on tile i
        sstore(buf ^ 1);           // stage tile i+1 into the other buffer
        buf ^= 1;
    }
    __syncthreads();
    compute(buf);                  // drain last tile

    // ---- epilogue (C/D layout: m = r + 8*(lane>>4), n = lane&15) ----
#pragma unroll
    for (int t = 0; t < 4; ++t) {
#pragma unroll
        for (int r = 0; r < 8; ++r) {
            int m = m0 + wm * 16 + r + 8 * hl;
            int n = n0 + wn * 64 + t * 16 + (lane & 15);
            if constexpr (MODE == 0) {
                float v = acc[t][r];
                if (bias) v += bias[n];
                C[(size_t)m * ldc + n] = v;
            } else if constexpr (MODE == 1) {
                float h1 = acc[t][r];
                C[(size_t)m * ldc + n] = h1 / (1.f + __expf(-h1)) * accB[t][r];
            } else {
                float sc = scale[(size_t)m * scale_stride];
                C[(size_t)m * ldc + n] += acc[t][r] * sc;
            }
        }
    }
}

// ---------------------------------------------------------------------------
// RMSNorm per row (D=1024)
// ---------------------------------------------------------------------------
__global__ __launch_bounds__(256)
void rmsnorm_k(const float* __restrict__ x, float* __restrict__ y)
{
    __shared__ float red[256];
    int row = blockIdx.x, tid = threadIdx.x;
    const float* xr = x + (size_t)row * DIM;
    float ss = 0.f;
    for (int c = tid; c < DIM; c += 256) { float v = xr[c]; ss += v * v; }
    red[tid] = ss; __syncthreads();
    for (int s = 128; s > 0; s >>= 1) {
        if (tid < s) red[tid] += red[tid + s];
        __syncthreads();
    }
    float scl = rsqrtf(red[0] / (float)DIM + 1e-6f);
    for (int c = tid; c < DIM; c += 256) y[(size_t)row * DIM + c] = xr[c] * scl;
}

// ---------------------------------------------------------------------------
// Householder Q build: column-independent sweep of 32 reflections.
// ---------------------------------------------------------------------------
__global__ void hh_kernel(const float* __restrict__ vs, float* __restrict__ Q)
{
    int j = threadIdx.x;     // column owner, 0..63
    float col[64];
#pragma unroll
    for (int i = 0; i < 64; ++i) col[i] = (i == j) ? 1.f : 0.f;
    for (int rI = 0; rI < 32; ++rI) {
        const float* v = vs + rI * 64;
        float vn = 1e-8f, wj = 0.f;
#pragma unroll
        for (int i = 0; i < 64; ++i) { float vi = v[i]; vn += vi * vi; wj += vi * col[i]; }
        float c = 2.f * wj / vn;
#pragma unroll
        for (int i = 0; i < 64; ++i) col[i] -= c * v[i];
    }
#pragma unroll
    for (int i = 0; i < 64; ++i) Q[i * 64 + j] = col[i];
}

// ---------------------------------------------------------------------------
// RnRoPE: row -> (r @ Q^T) -> RoPE rotate -> (@ Q). In place on qkv.
// ---------------------------------------------------------------------------
__global__ __launch_bounds__(256)
void rope_kernel(float* __restrict__ qkv, const float* __restrict__ Q,
                 const float* __restrict__ rope_pos, const float* __restrict__ inv_freq)
{
    __shared__ float Qs[64 * 64];
    __shared__ float rs[4][64];
    __shared__ float ts[4][64];
    __shared__ float tps[4][64];
    int tid = threadIdx.x;
    for (int i = tid; i < 4096; i += 256) Qs[i] = Q[i];
    int grp = tid >> 6, l = tid & 63;
    int rowid = blockIdx.x * 4 + grp;
    int which = rowid >> 15;            // 0 = q, 1 = k   (SEQ*NH = 32768)
    int rem   = rowid & 32767;
    int h = rem >> 11;
    int s = rem & 2047;
    float* ptr = qkv + (size_t)s * (3 * DIM) + which * DIM + h * HD;
    rs[grp][l] = ptr[l];
    __syncthreads();
    float t = 0.f;
#pragma unroll
    for (int i = 0; i < 64; ++i) t += rs[grp][i] * Qs[l * 64 + i];   // (r @ Q^T)[l]
    ts[grp][l] = t;
    int i2 = l & 31;
    float f = 0.f;
    if (i2 < 15) f = rope_pos[s * 3 + (i2 / 5)] * inv_freq[i2 % 5];
    float cv = cosf(f), sv = sinf(f);
    __syncthreads();
    float other = (l < 32) ? -ts[grp][l + 32] : ts[grp][l - 32];
    tps[grp][l] = t * cv + other * sv;
    __syncthreads();
    float o = 0.f;
#pragma unroll
    for (int i = 0; i < 64; ++i) o += tps[grp][i] * Qs[i * 64 + l];  // (t' @ Q)[l]
    ptr[l] = o;
}

// ---------------------------------------------------------------------------
// Online-softmax attention. 1 thread = 1 (head, q-row); K/V tiles in LDS.
// ---------------------------------------------------------------------------
__global__ __launch_bounds__(128)
void attn_kernel(const float* __restrict__ qkv, float* __restrict__ out)
{
    __shared__ float Ks[32][64];
    __shared__ float Vs[32][64];
    int h  = blockIdx.y;
    int qr = blockIdx.x * 128 + threadIdx.x;
    float q[64], o[64];
    const float* qp = qkv + (size_t)qr * (3 * DIM) + h * HD;
#pragma unroll
    for (int d = 0; d < 64; ++d) { q[d] = qp[d] * 0.125f; o[d] = 0.f; }
    float mmax = -1e30f, l = 0.f;
    for (int kt = 0; kt < SEQ; kt += 32) {
        __syncthreads();
#pragma unroll
        for (int i = 0; i < 16; ++i) {
            int idx = i * 128 + threadIdx.x;
            int kr = idx >> 6, dc = idx & 63;
            Ks[kr][dc] = qkv[(size_t)(kt + kr) * (3 * DIM) + DIM     + h * HD + dc];
            Vs[kr][dc] = qkv[(size_t)(kt + kr) * (3 * DIM) + 2 * DIM + h * HD + dc];
        }
        __syncthreads();
        float p[32];
        float tmax = mmax;
#pragma unroll
        for (int kk = 0; kk < 32; ++kk) {
            float sc = 0.f;
#pragma unroll
            for (int d = 0; d < 64; ++d) sc += q[d] * Ks[kk][d];
            p[kk] = sc;
            tmax = fmaxf(tmax, sc);
        }
        float corr = __expf(mmax - tmax);
        mmax = tmax;
        l *= corr;
#pragma unroll
        for (int d = 0; d < 64; ++d) o[d] *= corr;
#pragma unroll
        for (int kk = 0; kk < 32; ++kk) {
            float e = __expf(p[kk] - mmax);
            l += e;
#pragma unroll
            for (int d = 0; d < 64; ++d) o[d] += e * Vs[kk][d];
        }
    }
    float inv = 1.f / l;
    float* op = out + (size_t)qr * DIM + h * HD;
#pragma unroll
    for (int d = 0; d < 64; ++d) op[d] = o[d] * inv;
}

// ---------------------------------------------------------------------------
// x2 = x + proj * sigmoid(gate)
// ---------------------------------------------------------------------------
__global__ void gate_combine_k(const float* __restrict__ x, const float* __restrict__ proj,
                               const float* __restrict__ gate, float* __restrict__ x2, int n)
{
    int i = blockIdx.x * 256 + threadIdx.x;
    if (i < n) x2[i] = x[i] + proj[i] / (1.f + __expf(-gate[i]));
}

// ---------------------------------------------------------------------------
// Router: sigmoid top-3 combine weights + aux loss atomicAdd.
// ---------------------------------------------------------------------------
__global__ __launch_bounds__(256)
void router_kernel(const float* __restrict__ xn2, const float* __restrict__ rw,
                   const float* __restrict__ rb, float* __restrict__ combine,
                   float* __restrict__ aux)
{
    __shared__ float red[256];
    __shared__ float logits[NE];
    int s = blockIdx.x, tid = threadIdx.x;
    float part[NE] = {};
    for (int c = tid; c < DIM; c += 256) {
        float xv = xn2[(size_t)s * DIM + c];
#pragma unroll
        for (int e = 0; e < NE; ++e) part[e] += xv * rw[c * NE + e];
    }
    for (int e = 0; e < NE; ++e) {
        red[tid] = part[e]; __syncthreads();
        for (int st = 128; st > 0; st >>= 1) {
            if (tid < st) red[tid] += red[tid + st];
            __syncthreads();
        }
        if (tid == 0) logits[e] = red[0] + rb[e];
        __syncthreads();
    }
    if (tid == 0) {
        float sc[NE]; float auxs = 0.f;
        for (int e = 0; e < NE; ++e) {
            auxs += logits[e] * logits[e];
            sc[e] = 1.f / (1.f + __expf(-logits[e]));
        }
        atomicAdd(aux, 0.01f * auxs / (float)(SEQ * NE));
        bool used[NE] = {};
        int sel[3]; float sv[3]; float tot = 1e-6f;
        for (int k = 0; k < 3; ++k) {
            int best = 0; float bv = -1e30f;
            for (int e = 0; e < NE; ++e)
                if (!used[e] && sc[e] > bv) { bv = sc[e]; best = e; }
            used[best] = true; sel[k] = best; sv[k] = bv; tot += bv;
        }
        for (int e = 0; e < NE; ++e) combine[s * NE + e] = 0.f;
        for (int k = 0; k < 3; ++k) combine[s * NE + sel[k]] = sv[k] / tot;
    }
}

__global__ void zero1_k(float* p) { *p = 0.f; }

// ---------------------------------------------------------------------------
extern "C" void kernel_launch(void* const* d_in, const int* in_sizes, int n_in,
                              void* d_out, int out_size, void* d_ws, size_t ws_size,
                              hipStream_t stream)
{
    const float* x        = (const float*)d_in[0];
    const float* rope_pos = (const float*)d_in[1];
    const float* inv_freq = (const float*)d_in[2];
    const float* qkv_w    = (const float*)d_in[3];
    const float* hh_vs    = (const float*)d_in[4];
    const float* out_w    = (const float*)d_in[5];
    const float* gate_w   = (const float*)d_in[6];
    const float* gate_b   = (const float*)d_in[7];
    const float* router_w = (const float*)d_in[8];
    const float* router_b = (const float*)d_in[9];
    const float* w12      = (const float*)d_in[10];
    const float* w3       = (const float*)d_in[11];
    float* out = (float*)d_out;

    const size_t SD = (size_t)SEQ * DIM;
    float* ws = (float*)d_ws;
    size_t off = 0;
    float* xn      = ws + off; off += SD;
    float* qkvb    = ws + off; off += (size_t)SEQ * 3 * DIM;
    float* Qm      = ws + off; off += 64 * 64;
    float* attnout = ws + off; off += SD;
    float* proj    = ws + off; off += SD;
    float* gate    = ws + off; off += SD;
    float* x2      = ws + off; off += SD;
    float* xn2     = ws + off; off += SD;
    float* combine = ws + off; off += (size_t)SEQ * NE;
    float* G       = ws + off; off += (size_t)SEQ * HID;

    zero1_k<<<1, 1, 0, stream>>>(out + SD);                       // aux = 0

    // ---- attention path ----
    rmsnorm_k<<<SEQ, 256, 0, stream>>>(x, xn);
    hh_kernel<<<1, 64, 0, stream>>>(hh_vs, Qm);
    gemm_bf16wmma<0><<<dim3(3 * DIM / 128, SEQ / 64), 256, 0, stream>>>(
        xn, qkv_w, qkvb, nullptr, nullptr, 0, SEQ, 3 * DIM, DIM, DIM, 3 * DIM, 3 * DIM, 0);
    rope_kernel<<<SEQ * NH * 2 / 4, 256, 0, stream>>>(qkvb, Qm, rope_pos, inv_freq);
    attn_kernel<<<dim3(SEQ / 128, NH), 128, 0, stream>>>(qkvb, attnout);
    gemm_bf16wmma<0><<<dim3(DIM / 128, SEQ / 64), 256, 0, stream>>>(
        attnout, out_w, proj, nullptr, nullptr, 0, SEQ, DIM, DIM, DIM, DIM, DIM, 0);
    gemm_bf16wmma<0><<<dim3(DIM / 128, SEQ / 64), 256, 0, stream>>>(
        proj, gate_w, gate, gate_b, nullptr, 0, SEQ, DIM, DIM, DIM, DIM, DIM, 0);
    gate_combine_k<<<(int)((SD + 255) / 256), 256, 0, stream>>>(x, proj, gate, x2, (int)SD);

    // ---- MoE FFN path ----
    rmsnorm_k<<<SEQ, 256, 0, stream>>>(x2, xn2);
    router_kernel<<<SEQ, 256, 0, stream>>>(xn2, router_w, router_b, combine, out + SD);
    hipMemcpyAsync(out, x2, SD * sizeof(float), hipMemcpyDeviceToDevice, stream);  // out = resid

    for (int e = 0; e < NE; ++e) {
        const float* w12e = w12 + (size_t)e * DIM * 2 * HID;   // [DIM, 2*HID]
        const float* w3e  = w3  + (size_t)e * HID * DIM;       // [HID, DIM]
        // G = silu(xn2 @ w12e[:, :HID]) * (xn2 @ w12e[:, HID:])
        gemm_bf16wmma<1><<<dim3(HID / 128, SEQ / 64), 256, 0, stream>>>(
            xn2, w12e, G, nullptr, nullptr, 0, SEQ, HID, DIM, DIM, 2 * HID, HID, HID);
        // out += (G @ w3e) * combine[:, e]
        gemm_bf16wmma<2><<<dim3(DIM / 128, SEQ / 64), 256, 0, stream>>>(
            G, w3e, out, nullptr, combine + e, NE, SEQ, DIM, HID, HID, DIM, DIM, 0);
    }
}
